// CrossAttentionBlock_89232240542672
// MI455X (gfx1250) — matrix operand order repaired and checked by reading, not compiled, not run on hardware
//
#include <hip/hip_runtime.h>

// ---------------------------------------------------------------------------
// CDNA5 (gfx1250) CrossAttentionBlock: all GEMMs via v_wmma_f32_16x16x32_bf16
// K-loop double-buffered in registers: loads for step t+1 issued before the
// 8 chained WMMAs of step t, so s_wait_loadcnt lands behind ~8 WMMAs of cover.
// ---------------------------------------------------------------------------

typedef __attribute__((ext_vector_type(16))) __bf16          v16bf;
typedef __attribute__((ext_vector_type(8)))  __bf16          v8bf;
typedef __attribute__((ext_vector_type(8)))  float           v8f;
typedef __attribute__((ext_vector_type(8)))  unsigned short  v8us;

__device__ __forceinline__ unsigned short f2bf(float f) {
  unsigned u = __float_as_uint(f);
  u += 0x7fffu + ((u >> 16) & 1u);           // round-to-nearest-even
  return (unsigned short)(u >> 16);
}
__device__ __forceinline__ float bf2f(unsigned short h) {
  return __uint_as_float(((unsigned)h) << 16);
}

// problem dims
static constexpr int Bn = 4, Cn = 256, Hn = 128, Wn = 128;
static constexpr int HWn = Hn * Wn;          // 16384
static constexpr int HEADSn = 2, CHn = 128;  // per-head channels
static constexpr int HIDn = 680, HID2n = 1360, HIDPn = 704; // padded K for conv2

// ---------------------------------------------------------------------------
// Generic bf16 WMMA GEMM: D = A(MxK) * B(NxK, K-contig per row) [+bias][+res]
// block = 256 threads (8 waves); tile 128(M) x 128(N); K step 64 (2x32),
// double-buffered fragments. Requires K % 64 == 0 (true for 256/16384/128/704).
// mode: 0 = f32 scatter out[m*ldo+n], 1 = bf16 scatter, 2 = bf16 transposed
//       packed out[n*ldo + m] (16B stores), optional per-head split for vT.
// ---------------------------------------------------------------------------
__global__ __launch_bounds__(256) void gemm_bf16_wmma(
    const unsigned short* __restrict__ A, long long a_bs, int lda,
    const unsigned short* __restrict__ Bm, long long b_bs, int ldb,
    int M, int K,
    int mode, int headsplit, long long hs_stride,
    int osplitz, long long osplit_off,
    float* __restrict__ Of, unsigned short* __restrict__ Ob,
    long long o_bs, int ldo,
    const float* __restrict__ bias,
    const float* __restrict__ res, long long r_bs,
    const unsigned short* __restrict__ res2, long long r2_bs, int ldr2)
{
  const int z    = blockIdx.z;
  const int lane = threadIdx.x & 31;
  const int wv   = threadIdx.x >> 5;
  const int m_wave = blockIdx.y * 128 + wv * 16;
  if (m_wave >= M) return;                    // wave-uniform, no barriers used
  const int n0  = blockIdx.x * 128;
  const int l15 = lane & 15;
  const int hh  = lane >> 4;

  const unsigned short* Ap = A + (long long)z * a_bs + (long long)(m_wave + l15) * lda;
  const unsigned short* BpBase = Bm + (long long)z * b_bs;
  const unsigned short* Brow[8];
#pragma unroll
  for (int j2 = 0; j2 < 8; ++j2)
    Brow[j2] = BpBase + (long long)(n0 + j2 * 16 + l15) * ldb + hh * 16;

  v8f zero = {0.f,0.f,0.f,0.f,0.f,0.f,0.f,0.f};
  v8f acc[8];
#pragma unroll
  for (int j2 = 0; j2 < 8; ++j2) acc[j2] = zero;

  // A fragment (16x32 bf16): lane holds K = hh*16 + 0..15 (contiguous)
  auto ldA = [&](int k0, v16bf& a) {
    v8bf lo = *reinterpret_cast<const v8bf*>(Ap + k0 + hh * 16);
    v8bf hi = *reinterpret_cast<const v8bf*>(Ap + k0 + hh * 16 + 8);
#pragma unroll
    for (int i = 0; i < 8; ++i) { a[i] = lo[i]; a[i + 8] = hi[i]; }
  };
  // B fragments (32x16 bf16): lane n = l15, K = hh*16 + 0..15 (contiguous)
  auto ldB = [&](int k0, v16bf (&bb)[8]) {
#pragma unroll
    for (int j2 = 0; j2 < 8; ++j2) {
      v8bf lo = *reinterpret_cast<const v8bf*>(Brow[j2] + k0);
      v8bf hi = *reinterpret_cast<const v8bf*>(Brow[j2] + k0 + 8);
#pragma unroll
      for (int i = 0; i < 8; ++i) { bb[j2][i] = lo[i]; bb[j2][i + 8] = hi[i]; }
    }
  };
  auto mm = [&](const v16bf& a, const v16bf (&bb)[8]) {
#pragma unroll
    for (int j2 = 0; j2 < 8; ++j2)
      acc[j2] = __builtin_amdgcn_wmma_f32_16x16x32_bf16(
          false, a, false, bb[j2], (short)0, acc[j2], false, false);
  };

  v16bf a0, a1;
  v16bf b0[8], b1[8];
  ldA(0, a0); ldB(0, b0);
  for (int k0 = 0; k0 < K; k0 += 64) {
    __builtin_prefetch(Ap + k0 + 160, 0, 1);   // global_prefetch on A stream
    ldA(k0 + 32, a1); ldB(k0 + 32, b1);        // stage t+1 (overlaps wmma t)
    mm(a0, b0);
    if (k0 + 64 < K) { ldA(k0 + 64, a0); ldB(k0 + 64, b0); }  // stage t+2
    mm(a1, b1);
  }

  // ---- store -------------------------------------------------------------
  const int mstart = m_wave + 8 * hh;         // D rows m = mstart + j (contig)
  const long long ob = osplitz
      ? ((long long)(z >> 1) * o_bs + (long long)(z & 1) * osplit_off)
      : (long long)z * o_bs;

#pragma unroll
  for (int j2 = 0; j2 < 8; ++j2) {
    const int n = n0 + j2 * 16 + l15;
    if (mode == 2) {
      v8us pk;
#pragma unroll
      for (int j = 0; j < 8; ++j) {
        float v = acc[j2][j];
        if (bias) v += bias[mstart + j];
        pk[j] = f2bf(v);
      }
      long long oi;
      if (headsplit)
        oi = ob + (long long)(mstart >> 7) * hs_stride +
             (long long)n * ldo + (mstart & 127);
      else
        oi = ob + (long long)n * ldo + mstart;
      *reinterpret_cast<v8us*>(Ob + oi) = pk;
    } else {
#pragma unroll
      for (int j = 0; j < 8; ++j) {
        const int m = mstart + j;
        float v = acc[j2][j];
        if (bias) v += bias[m];
        if (res)  v += res[(long long)z * r_bs + (long long)m * ldo + n];
        if (res2) v += bf2f(res2[(long long)z * r2_bs + (long long)n * ldr2 + m]);
        const long long oi = ob + (long long)m * ldo + n;
        if (mode == 0) Of[oi] = v;
        else           Ob[oi] = f2bf(v);
      }
    }
  }
}

// ---------------------------------------------------------------------------
// Channel LayerNorm (over C=256) -> bf16 transposed [b, hw, c]
// ---------------------------------------------------------------------------
__global__ __launch_bounds__(256) void ln_to_bf16T(
    const float* __restrict__ X, const float* __restrict__ g,
    const float* __restrict__ bt, unsigned short* __restrict__ OT)
{
  const int idx = blockIdx.x * 256 + threadIdx.x;   // over B*HW = 65536
  const int b  = idx >> 14;
  const int hw = idx & (HWn - 1);
  const float* xp = X + b * Cn * HWn + hw;
  float s = 0.f, ss = 0.f;
#pragma unroll 4
  for (int c = 0; c < Cn; ++c) { float v = xp[c * HWn]; s += v; ss += v * v; }
  const float mean = s * (1.f / Cn);
  const float var  = ss * (1.f / Cn) - mean * mean;
  const float inv  = rsqrtf(var + 1e-6f);
  unsigned short* op = OT + (long long)idx * Cn;
#pragma unroll 4
  for (int c = 0; c < Cn; ++c)
    op[c] = f2bf((xp[c * HWn] - mean) * inv * g[c] + bt[c]);
}

// ---------------------------------------------------------------------------
// L2 normalize over per-head channels (128), f32 [z,c,d] -> bf16 same layout
// ---------------------------------------------------------------------------
__global__ __launch_bounds__(256) void l2norm_to_bf16(
    const float* __restrict__ Q, unsigned short* __restrict__ Oq)
{
  const int idx = blockIdx.x * 256 + threadIdx.x;   // over B*HEADS*HW = 131072
  const int z = idx >> 14;
  const int d = idx & (HWn - 1);
  const float* qp = Q + z * (CHn * HWn) + d;
  float ss = 0.f;
#pragma unroll 4
  for (int c = 0; c < CHn; ++c) { float v = qp[c * HWn]; ss += v * v; }
  const float inv = 1.f / fmaxf(sqrtf(ss), 1e-12f);
  unsigned short* op = Oq + z * (CHn * HWn) + d;
#pragma unroll 4
  for (int c = 0; c < CHn; ++c) op[c * HWn] = f2bf(qp[c * HWn] * inv);
}

// ---------------------------------------------------------------------------
// Softmax over rows of 128 with scale = 0.1*sigmoid(temperature[head])
// ---------------------------------------------------------------------------
__global__ __launch_bounds__(128) void softmax_scale(
    const float* __restrict__ af, const float* __restrict__ temp,
    unsigned short* __restrict__ ab)
{
  __shared__ float red[128];
  const int row = blockIdx.x;                  // B*HEADS*128 rows
  const int tid = threadIdx.x;
  const int h = (row >> 7) & 1;
  const float scale = 0.1f / (1.f + expf(-temp[h]));
  const float v = af[row * 128 + tid] * scale;
  red[tid] = v; __syncthreads();
  for (int s = 64; s > 0; s >>= 1) {
    if (tid < s) red[tid] = fmaxf(red[tid], red[tid + s]);
    __syncthreads();
  }
  const float mx = red[0]; __syncthreads();
  const float e = expf(v - mx);
  red[tid] = e; __syncthreads();
  for (int s = 64; s > 0; s >>= 1) {
    if (tid < s) red[tid] += red[tid + s];
    __syncthreads();
  }
  ab[row * 128 + tid] = f2bf(e / red[0]);
}

// ---------------------------------------------------------------------------
// Depthwise 3x3 (SAME) + tanh-GELU gate -> bf16 [b, hw, hidp] (K-contig)
// ---------------------------------------------------------------------------
__global__ __launch_bounds__(256) void dw_gelu_gate(
    const unsigned short* __restrict__ c1, const float* __restrict__ dww,
    const float* __restrict__ dwb, unsigned short* __restrict__ gate)
{
  const long long idx = (long long)blockIdx.x * 256 + threadIdx.x;
  if (idx >= (long long)Bn * HIDn * HWn) return;
  const int hw  = (int)(idx % HWn);
  const long long t = idx / HWn;
  const int hid = (int)(t % HIDn);
  const int b   = (int)(t / HIDn);
  const int yy = hw >> 7, xx = hw & 127;
  const unsigned short* p1 = c1 + (long long)(b * HID2n + hid) * HWn;
  const unsigned short* p2 = c1 + (long long)(b * HID2n + hid + HIDn) * HWn;
  const float* w1 = dww + hid * 9;
  const float* w2 = dww + (hid + HIDn) * 9;
  float a1 = 0.f, a2 = 0.f;
#pragma unroll
  for (int dy = -1; dy <= 1; ++dy) {
    const int y2 = yy + dy;
    if ((unsigned)y2 >= (unsigned)Hn) continue;
#pragma unroll
    for (int dx = -1; dx <= 1; ++dx) {
      const int x2 = xx + dx;
      if ((unsigned)x2 >= (unsigned)Wn) continue;
      const int o  = y2 * Wn + x2;
      const int wi = (dy + 1) * 3 + (dx + 1);
      a1 += w1[wi] * bf2f(p1[o]);
      a2 += w2[wi] * bf2f(p2[o]);
    }
  }
  a1 += dwb[hid];
  a2 += dwb[hid + HIDn];
  const float u = a1;
  const float gelu =
      0.5f * u * (1.f + tanhf(0.7978845608028654f * (u + 0.044715f * u * u * u)));
  gate[(long long)(b * HWn + hw) * HIDPn + hid] = f2bf(gelu * a2);
}

__global__ __launch_bounds__(256) void gate_pad_zero(unsigned short* __restrict__ gate)
{
  const int idx = blockIdx.x * 256 + threadIdx.x;   // B*HW*24
  if (idx >= Bn * HWn * (HIDPn - HIDn)) return;
  const int c = HIDn + idx % (HIDPn - HIDn);
  const int bhw = idx / (HIDPn - HIDn);
  gate[(long long)bhw * HIDPn + c] = 0;
}

// ---------------------------------------------------------------------------
// Weight conversion to bf16 (+ conv2 K-padding 680->704)
// ---------------------------------------------------------------------------
__global__ __launch_bounds__(256) void cvt_bf16(
    const float* __restrict__ s, unsigned short* __restrict__ d, int n)
{
  const int i = blockIdx.x * 256 + threadIdx.x;
  if (i < n) d[i] = f2bf(s[i]);
}
__global__ __launch_bounds__(256) void cvt_w2_pad(
    const float* __restrict__ s, unsigned short* __restrict__ d)
{
  const int i = blockIdx.x * 256 + threadIdx.x;   // 256*704
  if (i >= Cn * HIDPn) return;
  const int o = i / HIDPn, c = i % HIDPn;
  d[i] = (c < HIDn) ? f2bf(s[o * HIDn + c]) : (unsigned short)0;
}

// ---------------------------------------------------------------------------
// Host side
// ---------------------------------------------------------------------------
extern "C" void kernel_launch(void* const* d_in, const int* in_sizes, int n_in,
                              void* d_out, int out_size, void* d_ws, size_t ws_size,
                              hipStream_t stream) {
  (void)in_sizes; (void)n_in; (void)out_size; (void)ws_size;
  const float* x       = (const float*)d_in[0];
  const float* y       = (const float*)d_in[1];
  const float* n1w     = (const float*)d_in[2];
  const float* n1b     = (const float*)d_in[3];
  const float* q_w     = (const float*)d_in[4];
  const float* q_b     = (const float*)d_in[5];
  const float* kv_w    = (const float*)d_in[6];
  const float* kv_b    = (const float*)d_in[7];
  const float* proj_w  = (const float*)d_in[8];
  const float* proj_b  = (const float*)d_in[9];
  const float* temp    = (const float*)d_in[10];
  const float* n2w     = (const float*)d_in[11];
  const float* n2b     = (const float*)d_in[12];
  const float* conv1_w = (const float*)d_in[13];
  const float* conv1_b = (const float*)d_in[14];
  const float* dw_w    = (const float*)d_in[15];
  const float* dw_b    = (const float*)d_in[16];
  const float* conv2_w = (const float*)d_in[17];
  const float* conv2_b = (const float*)d_in[18];
  float* out = (float*)d_out;

  char* ws = (char*)d_ws;
  // workspace layout (bytes); a few regions reuse dead buffers
  const size_t o_wq    = 0;                         // 256*256 bf16
  const size_t o_wkv   = o_wq  + 131072;            // 512*256 bf16
  const size_t o_wpj   = o_wkv + 262144;            // 256*256 bf16
  const size_t o_w1    = o_wpj + 131072;            // 1360*256 bf16
  const size_t o_w2    = o_w1  + 696320;            // 256*704 bf16
  const size_t o_xlnT  = o_w2  + 360448;            // [b,hw,256] bf16 (later ln2T)
  const size_t o_ylnT  = o_xlnT + 33554432ull;      // [b,hw,256] bf16 (later outT)
  const size_t o_qf32  = o_ylnT + 33554432ull;      // [b,256,hw] f32  (later y1)
  const size_t o_kf32  = o_qf32 + 67108864ull;      // [b,256,hw] f32
  const size_t o_qhat  = o_kf32 + 67108864ull;      // bf16 [z,128,hw]
  const size_t o_khat  = o_qhat + 33554432ull;      // bf16 [z,128,hw]
  const size_t o_vT    = o_khat + 33554432ull;      // bf16 [z,hw,128]
  const size_t o_attnf = o_vT   + 33554432ull;      // f32  [z,128,128]
  const size_t o_attnb = o_attnf + 524288ull;       // bf16 [z,128,128]
  const size_t o_c1out = o_attnb + 262144ull;       // bf16 [b,1360,hw]
  const size_t o_gateT = o_c1out + 178257920ull;    // bf16 [b,hw,704]
  const size_t o_ln2T  = o_xlnT;                    // alias (xlnT dead)
  const size_t o_outT  = o_ylnT;                    // alias (ylnT dead)
  const size_t o_y1    = o_qf32;                    // alias (qf32 dead)

  unsigned short* wq   = (unsigned short*)(ws + o_wq);
  unsigned short* wkv  = (unsigned short*)(ws + o_wkv);
  unsigned short* wpj  = (unsigned short*)(ws + o_wpj);
  unsigned short* w1   = (unsigned short*)(ws + o_w1);
  unsigned short* w2   = (unsigned short*)(ws + o_w2);
  unsigned short* xlnT = (unsigned short*)(ws + o_xlnT);
  unsigned short* ylnT = (unsigned short*)(ws + o_ylnT);
  float*          qf32 = (float*)(ws + o_qf32);
  float*          kf32 = (float*)(ws + o_kf32);
  unsigned short* qhat = (unsigned short*)(ws + o_qhat);
  unsigned short* khat = (unsigned short*)(ws + o_khat);
  unsigned short* vT   = (unsigned short*)(ws + o_vT);
  float*          attnf= (float*)(ws + o_attnf);
  unsigned short* attnb= (unsigned short*)(ws + o_attnb);
  unsigned short* c1out= (unsigned short*)(ws + o_c1out);
  unsigned short* gateT= (unsigned short*)(ws + o_gateT);
  unsigned short* ln2T = (unsigned short*)(ws + o_ln2T);
  unsigned short* outT = (unsigned short*)(ws + o_outT);
  float*          y1   = (float*)(ws + o_y1);

  const dim3 blk256(256), blk128(128);
  const long long HW = HWn;

  // 1) weights -> bf16
  cvt_bf16<<<dim3(256), blk256, 0, stream>>>(q_w, wq, 256 * 256);
  cvt_bf16<<<dim3(512), blk256, 0, stream>>>(kv_w, wkv, 512 * 256);
  cvt_bf16<<<dim3(256), blk256, 0, stream>>>(proj_w, wpj, 256 * 256);
  cvt_bf16<<<dim3(1360), blk256, 0, stream>>>(conv1_w, w1, 1360 * 256);
  cvt_w2_pad<<<dim3((256 * HIDPn + 255) / 256), blk256, 0, stream>>>(conv2_w, w2);

  // 2) LayerNorm1 on x and y -> bf16 transposed
  ln_to_bf16T<<<dim3(256), blk256, 0, stream>>>(x, n1w, n1b, xlnT);
  ln_to_bf16T<<<dim3(256), blk256, 0, stream>>>(y, n1w, n1b, ylnT);

  // 3) q = q_w @ xln (f32 out for normalization)
  gemm_bf16_wmma<<<dim3(128, 2, 4), blk256, 0, stream>>>(
      wq, 0, 256, xlnT, HW * 256, 256, 256, 256,
      0, 0, 0, 0, 0, qf32, nullptr, 256 * HW, HWn,
      q_b, nullptr, 0, nullptr, 0, 0);
  // 4) k = kv_w[0:256] @ yln (f32)
  gemm_bf16_wmma<<<dim3(128, 2, 4), blk256, 0, stream>>>(
      wkv, 0, 256, ylnT, HW * 256, 256, 256, 256,
      0, 0, 0, 0, 0, kf32, nullptr, 256 * HW, HWn,
      kv_b, nullptr, 0, nullptr, 0, 0);
  // 5) v = kv_w[256:512] @ yln -> bf16 transposed [b,h,hw,128]
  gemm_bf16_wmma<<<dim3(128, 2, 4), blk256, 0, stream>>>(
      wkv + 256 * 256, 0, 256, ylnT, HW * 256, 256, 256, 256,
      2, 1, HW * 128, 0, 0, nullptr, vT, 2 * HW * 128, 128,
      kv_b + 256, nullptr, 0, nullptr, 0, 0);

  // 6) L2-normalize q,k over per-head channels -> bf16
  l2norm_to_bf16<<<dim3(512), blk256, 0, stream>>>(qf32, qhat);
  l2norm_to_bf16<<<dim3(512), blk256, 0, stream>>>(kf32, khat);

  // 7) attn = qhat @ khat^T over d=16384 (per b,head) -> f32 [z,128,128]
  gemm_bf16_wmma<<<dim3(1, 1, 8), blk256, 0, stream>>>(
      qhat, (long long)CHn * HW, HWn, khat, (long long)CHn * HW, HWn, 128, HWn,
      0, 0, 0, 0, 0, attnf, nullptr, 128 * 128, 128,
      nullptr, nullptr, 0, nullptr, 0, 0);

  // 8) softmax with 0.1*sigmoid(temperature) scale -> bf16
  softmax_scale<<<dim3(Bn * HEADSn * 128), blk128, 0, stream>>>(attnf, temp, attnb);

  // 9) out = attn @ v -> bf16 transposed [b,hw,256] (head folded into offset)
  gemm_bf16_wmma<<<dim3(128, 1, 8), blk256, 0, stream>>>(
      attnb, 128 * 128, 128, vT, HW * 128, 128, 128, 128,
      2, 0, 0, 1, 128, nullptr, outT, HW * 256, 256,
      nullptr, nullptr, 0, nullptr, 0, 0);

  // 10) y1 = y + proj_w @ out + proj_b  (f32 [b,256,hw])
  gemm_bf16_wmma<<<dim3(128, 2, 4), blk256, 0, stream>>>(
      wpj, 0, 256, outT, HW * 256, 256, 256, 256,
      0, 0, 0, 0, 0, y1, nullptr, 256 * HW, HWn,
      proj_b, y, 256 * HW, nullptr, 0, 0);

  // 11) LayerNorm2 on y1 -> bf16 transposed
  ln_to_bf16T<<<dim3(256), blk256, 0, stream>>>(y1, n2w, n2b, ln2T);

  // 12) conv1: 1360x256 @ ln2 -> bf16 [b,1360,hw]
  gemm_bf16_wmma<<<dim3(128, 11, 4), blk256, 0, stream>>>(
      w1, 0, 256, ln2T, HW * 256, 256, HID2n, 256,
      1, 0, 0, 0, 0, nullptr, c1out, (long long)HID2n * HW, HWn,
      conv1_b, nullptr, 0, nullptr, 0, 0);

  // 13) depthwise 3x3 + gelu gate -> bf16 [b,hw,704]; zero pad channels
  gate_pad_zero<<<dim3((Bn * HWn * 24 + 255) / 256), blk256, 0, stream>>>(gateT);
  {
    const long long tot = (long long)Bn * HIDn * HWn;
    dw_gelu_gate<<<dim3((unsigned)((tot + 255) / 256)), blk256, 0, stream>>>(
        c1out, dw_w, dw_b, gateT);
  }

  // 14) out = y1 + ln2(y1) + conv2_w @ gate + conv2_b  (inner ffn residual!)
  gemm_bf16_wmma<<<dim3(128, 2, 4), blk256, 0, stream>>>(
      w2, 0, HIDPn, gateT, HW * HIDPn, HIDPn, 256, HIDPn,
      0, 0, 0, 0, 0, out, nullptr, 256 * HW, HWn,
      conv2_b, y1, 256 * HW, ln2T, HW * 256, 256);
}
